// GODIVA_60885456388392
// MI455X (gfx1250) — matrix operand order repaired and checked by reading, not compiled, not run on hardware
//
#include <hip/hip_runtime.h>

// ---------------- CDNA5 / gfx1250 types & WMMA ----------------
typedef _Float16 f16;
typedef __attribute__((ext_vector_type(16))) _Float16 v16h;
typedef __attribute__((ext_vector_type(8)))  _Float16 v8h;
typedef __attribute__((ext_vector_type(8)))  float    v8f;

#define WMMA_F16(A, B, C) \
  __builtin_amdgcn_wmma_f32_16x16x32_f16(false, (A), false, (B), (short)0, (C), false, false)

// readfirstlane: values that are uniform per wave but derived from threadIdx
// (wave id) are otherwise treated as divergent -> VGPR math + EXEC-mask
// branches + spills. RFL pins them to SGPRs / SALU control flow.
#define RFL(x) __builtin_amdgcn_readfirstlane(x)

// Problem constants (match reference): B=8, S=1024, D=512, H=8, dh=64
// Token grid (4,16,16): t = q>>8, r = (q>>4)&15, c = q&15
// mask(q,k) = (0<=q-k<=16) || (r,c eq && tk<=tq) || (t,c eq && rk<=rq)

__device__ __forceinline__ bool mask_ok(int q, int k) {
  int d = q - k;
  if (d >= 0 && d <= 16) return true;
  int tq = q >> 8, rq = (q >> 4) & 15, cq = q & 15;
  int tk = k >> 8, rk = (k >> 4) & 15, ck = k & 15;
  if (rk == rq && ck == cq && tk <= tq) return true;
  if (tk == tq && ck == cq && rk <= rq) return true;
  return false;
}

// i-th 16-key block needed by tile starting at q0 (t = frame, r = row)
__device__ __forceinline__ int block_start(int i, int t, int r, int q0, int eb) {
  if (i < t) return q0 - 256 * (t - i);      // time-axial blocks (earlier frames)
  i -= t;
  if (i < eb) return q0 - 16;                // band spill when r==0
  return (q0 & ~255) + 16 * (i - eb);        // current frame rows 0..r
}

// ---------------- weight transpose + f16 convert: Wt[n][k] = (f16)W[k][n] ----
__global__ void k_wtrans(const float* __restrict__ W, f16* __restrict__ Wt) {
  int idx = blockIdx.x * 256 + threadIdx.x;  // 512*512 elems
  int k = idx >> 9, n = idx & 511;
  Wt[n * 512 + k] = (f16)W[idx];
}

// ---------------- GEMM: out[8192,512] = X[8192,512] @ W[512,512] + bias -----
// Wt is f16, transposed [n][k]. One wave -> 16x64 tile; block = 8 waves -> 128x64.
template <int A_F32, int OUT_F32, int TRANS_V>
__global__ void k_gemm(const void* __restrict__ Xv, const f16* __restrict__ Wt,
                       const float* __restrict__ bias, void* __restrict__ Ov,
                       float scale) {
  const int lane = threadIdx.x & 31;
  const int w    = RFL(threadIdx.x >> 5);        // uniform per wave
  const int col  = lane & 15;
  const int hf   = lane >> 4;
  const int bn   = blockIdx.x & 7;
  const int bm   = blockIdx.x >> 3;
  const int m0   = bm * 128 + w * 16;            // SGPR
  const int n0   = bn * 64;                      // SGPR
  const int row  = m0 + col;

  v8f acc[4] = {};
  const float* Xf = (const float*)Xv;
  const f16*   Xh = (const f16*)Xv;

  for (int k0 = 0; k0 < 512; k0 += 32) {
    v16h a;
    if (A_F32) {
      const float* p = Xf + (size_t)row * 512 + k0 + 16 * hf;
      v8f lo = *(const v8f*)p;
      v8f hi = *(const v8f*)(p + 8);
#pragma unroll
      for (int j = 0; j < 8; ++j) { a[j] = (f16)lo[j]; a[8 + j] = (f16)hi[j]; }
    } else {
      const f16* p = Xh + (size_t)row * 512 + k0 + 16 * hf;
      v8h lo = *(const v8h*)p;
      v8h hi = *(const v8h*)(p + 8);
#pragma unroll
      for (int j = 0; j < 8; ++j) { a[j] = lo[j]; a[8 + j] = hi[j]; }
    }
#pragma unroll
    for (int f = 0; f < 4; ++f) {
      const f16* bp = Wt + (size_t)(n0 + 16 * f + col) * 512 + k0 + 16 * hf;
      v16h bfr = *(const v16h*)bp;
      acc[f] = WMMA_F16(a, bfr, acc[f]);
    }
  }

#pragma unroll
  for (int f = 0; f < 4; ++f) {
    int n = n0 + 16 * f + col;
    float bb = bias[n];
#pragma unroll
    for (int v = 0; v < 8; ++v) {
      int rg = m0 + v + 8 * hf;
      float val = (acc[f][v] + bb) * scale;
      if (TRANS_V) {                        // Vt[(b*512 + n)*1024 + s]
        int b = rg >> 10, s = rg & 1023;
        ((f16*)Ov)[(size_t)(b * 512 + n) * 1024 + s] = (f16)val;
      } else if (OUT_F32) {
        ((float*)Ov)[(size_t)rg * 512 + n] = val;
      } else {
        ((f16*)Ov)[(size_t)rg * 512 + n] = (f16)val;
      }
    }
  }
}

// ---------------- sparse flash attention: one wave per (b,h,16-query tile) --
__global__ void k_attn(const f16* __restrict__ Qh, const f16* __restrict__ Kh,
                       const f16* __restrict__ Vt, f16* __restrict__ Ah) {
  __shared__ __align__(32) _Float16 pbuf[8][16 * 32];   // per-wave P tile

  const int lane = threadIdx.x & 31;
  const int w    = RFL(threadIdx.x >> 5);  // uniform per wave -> SGPR chain
  const int col  = lane & 15;
  const int hf   = lane >> 4;
  const int wg   = blockIdx.x * 8 + w;     // 4096 tiles total (all SGPR)
  const int qt   = wg & 63;
  const int h    = (wg >> 6) & 7;
  const int b    = wg >> 9;
  const int q0   = qt * 16;
  const int t    = q0 >> 8;
  const int r    = (q0 >> 4) & 15;

  const f16* Qb = Qh + (size_t)b * 1024 * 512 + h * 64;
  const f16* Kb = Kh + (size_t)b * 1024 * 512 + h * 64;
  const f16* Vb = Vt + (size_t)(b * 8 + h) * 64 * 1024;

  // Q A-fragments for dh=64 (two K=32 steps), loaded once
  v16h aq0, aq1;
  {
    const f16* qp = Qb + (size_t)(q0 + col) * 512;
    v8h l0 = *(const v8h*)(qp + 16 * hf);
    v8h h0 = *(const v8h*)(qp + 8 + 16 * hf);
    v8h l1 = *(const v8h*)(qp + 32 + 16 * hf);
    v8h h1 = *(const v8h*)(qp + 40 + 16 * hf);
#pragma unroll
    for (int j = 0; j < 8; ++j) {
      aq0[j] = l0[j]; aq0[8 + j] = h0[j];
      aq1[j] = l1[j]; aq1[8 + j] = h1[j];
    }
  }

  v8f o[4] = {};
  float m_run[8], l_run[8];
#pragma unroll
  for (int v = 0; v < 8; ++v) { m_run[v] = -1e30f; l_run[v] = 0.0f; }

  const int eb = (r == 0 && t > 0) ? 1 : 0;
  const int nb = t + eb + r + 1;

  for (int i = 0; i < nb; i += 2) {
    // wave-uniform block starts -> SALU + s_cbranch (EXEC stays all-ones)
    const int kbA = RFL(block_start(i, t, r, q0, eb));
    const int kbB = RFL((i + 1 < nb) ? block_start(i + 1, t, r, q0, eb) : -1);

    // ---- logits: S = Q (16x64) x K^T (64x16) per 16-key block
    v8f sA = {}, sB = {};
    {
      const f16* kp = Kb + (size_t)kbA * 512 + (size_t)col * 512 + 16 * hf;
      v16h kf0 = *(const v16h*)kp;
      v16h kf1 = *(const v16h*)(kp + 32);
      sA = WMMA_F16(aq0, kf0, sA);
      sA = WMMA_F16(aq1, kf1, sA);
    }
    if (kbB >= 0) {
      const f16* kp = Kb + (size_t)kbB * 512 + (size_t)col * 512 + 16 * hf;
      v16h kf0 = *(const v16h*)kp;
      v16h kf1 = *(const v16h*)(kp + 32);
      sB = WMMA_F16(aq0, kf0, sB);
      sB = WMMA_F16(aq1, kf1, sB);
    }

    // ---- mask + online softmax (rows m = v + 8*hf, key = kb + col)
    float lA[8], lB[8];
    const int kA = kbA + col;
    const int kB = (kbB >= 0 ? kbB : 0) + col;
#pragma unroll
    for (int v = 0; v < 8; ++v) {
      int q = q0 + v + 8 * hf;
      lA[v] = mask_ok(q, kA) ? sA[v] : -1e30f;
      lB[v] = (kbB >= 0 && mask_ok(q, kB)) ? sB[v] : -1e30f;
    }
    float mx[8];
#pragma unroll
    for (int v = 0; v < 8; ++v) mx[v] = fmaxf(lA[v], lB[v]);
#pragma unroll
    for (int off = 1; off < 16; off <<= 1)
#pragma unroll
      for (int v = 0; v < 8; ++v) mx[v] = fmaxf(mx[v], __shfl_xor(mx[v], off));

    float pA[8], pB[8], rs[8];
#pragma unroll
    for (int v = 0; v < 8; ++v) {
      float nm = fmaxf(m_run[v], mx[v]);
      float sc = __expf(m_run[v] - nm);
      pA[v] = __expf(lA[v] - nm);
      pB[v] = __expf(lB[v] - nm);
      rs[v] = pA[v] + pB[v];
      l_run[v] *= sc;
      m_run[v] = nm;
#pragma unroll
      for (int f = 0; f < 4; ++f) o[f][v] *= sc;
    }
#pragma unroll
    for (int off = 1; off < 16; off <<= 1)
#pragma unroll
      for (int v = 0; v < 8; ++v) rs[v] += __shfl_xor(rs[v], off);
#pragma unroll
    for (int v = 0; v < 8; ++v) l_run[v] += rs[v];

    // ---- D-layout P -> LDS -> A-layout fragment (16x32 f16)
#pragma unroll
    for (int v = 0; v < 8; ++v) {
      int m = v + 8 * hf;
      pbuf[w][m * 32 + col]      = (f16)pA[v];
      pbuf[w][m * 32 + 16 + col] = (f16)pB[v];   // zeros when kbB<0
    }
    __asm__ volatile("s_wait_dscnt 0" ::: "memory");
    v16h pf;
    {
      const _Float16* pp = &pbuf[w][col * 32 + 16 * hf];
      v8h lo = *(const v8h*)pp;
      v8h hi = *(const v8h*)(pp + 8);
#pragma unroll
      for (int j = 0; j < 8; ++j) { pf[j] = lo[j]; pf[8 + j] = hi[j]; }
    }

    // ---- O += P (16x32) x V (32x64): V B-frag contiguous from Vt[d][s]
    const int kbB_s = (kbB >= 0) ? kbB : kbA;    // uniform
    const int keyb  = hf ? kbB_s : kbA;          // per-lane select of two SGPRs
#pragma unroll
    for (int f = 0; f < 4; ++f) {
      v16h vf = *(const v16h*)(Vb + (size_t)(16 * f + col) * 1024 + keyb);
      o[f] = WMMA_F16(pf, vf, o[f]);
    }
  }

  // ---- epilogue: O / l, merge heads into Ah[b*1024+q][h*64+n] (f16)
#pragma unroll
  for (int f = 0; f < 4; ++f)
#pragma unroll
    for (int v = 0; v < 8; ++v) {
      int m = v + 8 * hf;
      float val = o[f][v] / l_run[v];
      Ah[(size_t)(b * 1024 + q0 + m) * 512 + h * 64 + 16 * f + col] = (f16)val;
    }
}

// ---------------- host driver ----------------
extern "C" void kernel_launch(void* const* d_in, const int* in_sizes, int n_in,
                              void* d_out, int out_size, void* d_ws, size_t ws_size,
                              hipStream_t stream) {
  (void)in_sizes; (void)n_in; (void)out_size; (void)ws_size;
  const float* query = (const float*)d_in[0];
  const float* key   = (const float*)d_in[1];
  const float* value = (const float*)d_in[2];
  const float* Wq    = (const float*)d_in[3];
  const float* bq    = (const float*)d_in[4];
  const float* Wk    = (const float*)d_in[5];
  const float* bk    = (const float*)d_in[6];
  const float* Wv    = (const float*)d_in[7];
  const float* bv    = (const float*)d_in[8];
  const float* Wo    = (const float*)d_in[9];
  const float* bo    = (const float*)d_in[10];

  char* ws = (char*)d_ws;
  const size_t WSZ = (size_t)512 * 512 * sizeof(f16);   // 512 KiB per weight
  const size_t TSZ = (size_t)8192 * 512 * sizeof(f16);  // 8 MiB per tensor
  f16* WtQ = (f16*)(ws + 0 * WSZ);
  f16* WtK = (f16*)(ws + 1 * WSZ);
  f16* WtV = (f16*)(ws + 2 * WSZ);
  f16* WtO = (f16*)(ws + 3 * WSZ);
  f16* Qh  = (f16*)(ws + 4 * WSZ + 0 * TSZ);
  f16* Kh  = (f16*)(ws + 4 * WSZ + 1 * TSZ);
  f16* Vt  = (f16*)(ws + 4 * WSZ + 2 * TSZ);
  f16* Ah  = (f16*)(ws + 4 * WSZ + 3 * TSZ);

  dim3 blk(256);
  k_wtrans<<<1024, blk, 0, stream>>>(Wq, WtQ);
  k_wtrans<<<1024, blk, 0, stream>>>(Wk, WtK);
  k_wtrans<<<1024, blk, 0, stream>>>(Wv, WtV);
  k_wtrans<<<1024, blk, 0, stream>>>(Wo, WtO);

  k_gemm<1, 0, 0><<<512, blk, 0, stream>>>(query, WtQ, bq, Qh, 0.125f); // 1/sqrt(64)
  k_gemm<1, 0, 0><<<512, blk, 0, stream>>>(key,   WtK, bk, Kh, 1.0f);
  k_gemm<1, 0, 1><<<512, blk, 0, stream>>>(value, WtV, bv, Vt, 1.0f);   // transposed

  k_attn<<<512, blk, 0, stream>>>(Qh, Kh, Vt, Ah);

  k_gemm<0, 1, 0><<<512, blk, 0, stream>>>(Ah, WtO, bo, d_out, 1.0f);
}
